// scaledDotProudct_31696858644695
// MI455X (gfx1250) — matrix operand order repaired and checked by reading, not compiled, not run on hardware
//
#include <hip/hip_runtime.h>
#include <math.h>

typedef __bf16 bf16;
typedef __attribute__((ext_vector_type(16))) __bf16 v16bf;
typedef __attribute__((ext_vector_type(8)))  __bf16 v8bf;
typedef __attribute__((ext_vector_type(4)))  __bf16 v4bf;
typedef __attribute__((ext_vector_type(8)))  float  v8f;

// ---------------------------------------------------------------------------
// Stage 0a: elementwise f32 -> bf16 convert (for x)
// ---------------------------------------------------------------------------
__global__ __launch_bounds__(256)
void convert_f32_bf16(const float* __restrict__ src, bf16* __restrict__ dst, int n) {
    int i = (blockIdx.x * 256 + threadIdx.x) * 4;
    if (i < n) {
        float4 f = *(const float4*)(src + i);
        v4bf o = { (bf16)f.x, (bf16)f.y, (bf16)f.z, (bf16)f.w };
        *(v4bf*)(dst + i) = o;
    }
}

// ---------------------------------------------------------------------------
// Stage 0b: 1024x1024 f32 -> bf16 transpose (q,k,v -> column-major B operands)
// ---------------------------------------------------------------------------
__global__ __launch_bounds__(256)
void convert_transpose_1024(const float* __restrict__ src, bf16* __restrict__ dst) {
    __shared__ float tile[32][33];
    const int bx = blockIdx.x * 32;
    const int by = blockIdx.y * 32;
    const int tx = threadIdx.x & 31;
    const int ty = threadIdx.x >> 5;
#pragma unroll
    for (int i = 0; i < 32; i += 8)
        tile[ty + i][tx] = src[(size_t)(by + ty + i) * 1024 + (bx + tx)];
    __syncthreads();
#pragma unroll
    for (int i = 0; i < 32; i += 8)
        dst[(size_t)(bx + ty + i) * 1024 + (by + tx)] = (bf16)tile[tx][ty + i];
}

// ---------------------------------------------------------------------------
// LDS double-buffered bf16 WMMA GEMM:
//   C[M,N] = scale * (A[M,K] @ B),  B given COLUMN-major (Bc[n*K+k] == B[k][n])
// Block: 256 threads = 8 waves, block tile 128x128, wave tile 32x64
// (2 A-frags x 4 B-frags = 8 WMMA accumulators per wave).
// K staged 32 at a time into LDS; A tile [128 rows x 64B], B tile
// [128 cols x 64B], both padded to 80B rows => conflict-free ds_load_b128.
// Fragment loads are hoisted into distinct registers so all ds_loads issue
// first, then 8 WMMAs fire back-to-back on the matrix pipe.
// Grid = (N/128, M/128).
// ---------------------------------------------------------------------------
enum { OUT_F32_ROW = 0, OUT_BF16_ROW = 1, OUT_BF16_TRANS = 2 };

template <int OUT_MODE>
__global__ __launch_bounds__(256)
void gemm_bf16_wmma(const bf16* __restrict__ A, const bf16* __restrict__ Bc,
                    void* __restrict__ Cout, int M, int N, int K,
                    int ldOut, float scale)
{
    constexpr int TK        = 32;   // k-depth per stage
    constexpr int ROW_BYTES = 80;   // 64B payload + 16B pad (20 dwords: 16 lanes -> 16 banks)
    // [buffer][ A: 128 rows | B: 128 cols ] * 80B
    __shared__ __align__(16) char lds[2][256 * ROW_BYTES];

    const int t    = threadIdx.x;
    const int lane = t & 31;
    const int wave = t >> 5;
    const int half = lane >> 4;
    const int l16  = lane & 15;
    const int waveM = wave >> 1;    // 0..3 -> 32-row strips
    const int waveN = wave & 1;     // 0..1 -> 64-col strips

    const int m0 = blockIdx.y * 128;
    const int n0 = blockIdx.x * 128;

    // Staging role: thread t fills row (t>>1), 32B chunk (t&1) of both tiles.
    const int sRow = t >> 1;
    const int sChk = t & 1;
    const bf16* gA = A  + (size_t)(m0 + sRow) * K + sChk * 16;
    const bf16* gB = Bc + (size_t)(n0 + sRow) * K + sChk * 16;
    const int sOff = sRow * ROW_BYTES + sChk * 32;

    // ---- prologue: stage k-tile 0 into buffer 0
    v8bf ra0 = *(const v8bf*)(gA);
    v8bf ra1 = *(const v8bf*)(gA + 8);
    v8bf rb0 = *(const v8bf*)(gB);
    v8bf rb1 = *(const v8bf*)(gB + 8);
    {
        char* dA = lds[0] + sOff;
        char* dB = lds[0] + 128 * ROW_BYTES + sOff;
        *(v8bf*)(dA)      = ra0;  *(v8bf*)(dA + 16) = ra1;
        *(v8bf*)(dB)      = rb0;  *(v8bf*)(dB + 16) = rb1;
    }
    __syncthreads();

    v8f acc[2][4] = {};
    int cur = 0;

    for (int kk = 0; kk < K; kk += TK) {
        const bool hasNext = (kk + TK) < K;
        if (hasNext) {   // prefetch next k-tile to registers (overlaps WMMA below)
            ra0 = *(const v8bf*)(gA + kk + TK);
            ra1 = *(const v8bf*)(gA + kk + TK + 8);
            rb0 = *(const v8bf*)(gB + kk + TK);
            rb1 = *(const v8bf*)(gB + kk + TK + 8);
        }

        const char* aB = lds[cur];
        const char* bB = lds[cur] + 128 * ROW_BYTES;

        // ---- load phase: all fragments into distinct registers first ----
        // A fragments: lane holds row, chunks at bytes 16*half and 32+16*half
        v16bf af[2];
#pragma unroll
        for (int f = 0; f < 2; ++f) {
            const int row = waveM * 32 + f * 16 + l16;
            const char* p = aB + row * ROW_BYTES + half * 16;
            v8bf lo = *(const v8bf*)(p);
            v8bf hi = *(const v8bf*)(p + 32);
#pragma unroll
            for (int i = 0; i < 8; ++i) { af[f][i] = lo[i]; af[f][8 + i] = hi[i]; }
        }
        // B fragments: lane holds column, chunks at bytes 16*half and 32+16*half
        v16bf bfr[4];
#pragma unroll
        for (int j = 0; j < 4; ++j) {
            const int col = waveN * 64 + j * 16 + l16;
            const char* p = bB + col * ROW_BYTES + half * 16;
            v8bf b0 = *(const v8bf*)(p);
            v8bf b1 = *(const v8bf*)(p + 32);
#pragma unroll
            for (int i = 0; i < 8; ++i) { bfr[j][i] = b0[i]; bfr[j][8 + i] = b1[i]; }
        }

        // ---- compute phase: 8 back-to-back WMMAs on the matrix pipe ----
#pragma unroll
        for (int j = 0; j < 4; ++j) {
            acc[0][j] = __builtin_amdgcn_wmma_f32_16x16x32_bf16(
                false, af[0], false, bfr[j], (short)0, acc[0][j], false, false);
            acc[1][j] = __builtin_amdgcn_wmma_f32_16x16x32_bf16(
                false, af[1], false, bfr[j], (short)0, acc[1][j], false, false);
        }

        if (hasNext) {   // write prefetched tile into the other buffer
            char* dA = lds[cur ^ 1] + sOff;
            char* dB = lds[cur ^ 1] + 128 * ROW_BYTES + sOff;
            *(v8bf*)(dA)      = ra0;  *(v8bf*)(dA + 16) = ra1;
            *(v8bf*)(dB)      = rb0;  *(v8bf*)(dB + 16) = rb1;
        }
        __syncthreads();
        cur ^= 1;
    }

    // D layout: VGPR i -> row = base + 8*half + i, col = base + l16
#pragma unroll
    for (int f = 0; f < 2; ++f) {
#pragma unroll
        for (int j = 0; j < 4; ++j) {
            const int col = n0 + waveN * 64 + 16 * j + l16;
#pragma unroll
            for (int i = 0; i < 8; ++i) {
                const int row = m0 + waveM * 32 + f * 16 + 8 * half + i;
                const float val = acc[f][j][i] * scale;
                if (OUT_MODE == OUT_F32_ROW) {
                    ((float*)Cout)[(size_t)row * ldOut + col] = val;
                } else if (OUT_MODE == OUT_BF16_ROW) {
                    ((bf16*)Cout)[(size_t)row * ldOut + col] = (bf16)val;
                } else { // OUT_BF16_TRANS
                    ((bf16*)Cout)[(size_t)col * ldOut + row] = (bf16)val;
                }
            }
        }
    }
}

// ---------------------------------------------------------------------------
// Row softmax over 1024 columns, in place. One 256-thread block per row.
// ---------------------------------------------------------------------------
__global__ __launch_bounds__(256)
void softmax_rows_1024(float* __restrict__ data) {
    __shared__ float red[256];
    float* row = data + (size_t)blockIdx.x * 1024;
    const int t = threadIdx.x;

    float v[4];
    float m = -INFINITY;
#pragma unroll
    for (int i = 0; i < 4; ++i) { v[i] = row[t + 256 * i]; m = fmaxf(m, v[i]); }
    red[t] = m; __syncthreads();
    for (int s = 128; s > 0; s >>= 1) {
        if (t < s) red[t] = fmaxf(red[t], red[t + s]);
        __syncthreads();
    }
    m = red[0]; __syncthreads();

    float sum = 0.f;
#pragma unroll
    for (int i = 0; i < 4; ++i) { v[i] = __expf(v[i] - m); sum += v[i]; }
    red[t] = sum; __syncthreads();
    for (int s = 128; s > 0; s >>= 1) {
        if (t < s) red[t] += red[t + s];
        __syncthreads();
    }
    const float inv = 1.0f / red[0];
#pragma unroll
    for (int i = 0; i < 4; ++i) row[t + 256 * i] = v[i] * inv;
}

// ---------------------------------------------------------------------------
// Orchestration
// ---------------------------------------------------------------------------
extern "C" void kernel_launch(void* const* d_in, const int* in_sizes, int n_in,
                              void* d_out, int out_size, void* d_ws, size_t ws_size,
                              hipStream_t stream) {
    (void)in_sizes; (void)n_in; (void)out_size; (void)ws_size;
    const float* x = (const float*)d_in[0];
    const float* q = (const float*)d_in[1];
    const float* k = (const float*)d_in[2];
    const float* v = (const float*)d_in[3];
    float* out = (float*)d_out;

    constexpr int N = 4096, D = 1024;
    char* ws = (char*)d_ws;
    bf16* xb  = (bf16*)(ws);                            //  8 MiB  x bf16        [N x D]
    bf16* qT  = (bf16*)(ws + ( 8u << 20));              //  2 MiB  q^T bf16      [D x D]
    bf16* kT  = (bf16*)(ws + (10u << 20));              //  2 MiB  k^T bf16
    bf16* vT  = (bf16*)(ws + (12u << 20));              //  2 MiB  v^T bf16
    bf16* qmb = (bf16*)(ws + (14u << 20));              //  8 MiB  qm bf16 row   [N x D]
    bf16* kmb = (bf16*)(ws + (22u << 20));              //  8 MiB  km bf16 row   [N x D]
    bf16* vmT = (bf16*)(ws + (30u << 20));              //  8 MiB  vm^T bf16     [D x N]
    bf16* sb  = (bf16*)(ws + (38u << 20));              // 32 MiB  scores bf16   [N x N]

    // Stage 0: precision conversion / operand layout
    convert_f32_bf16<<<(N * D) / 1024, 256, 0, stream>>>(x, xb, N * D);
    dim3 tg(32, 32);
    convert_transpose_1024<<<tg, 256, 0, stream>>>(q, qT);
    convert_transpose_1024<<<tg, 256, 0, stream>>>(k, kT);
    convert_transpose_1024<<<tg, 256, 0, stream>>>(v, vT);

    // Stage 1: projections. qm,km row-major (km row-major == (km.T) col-major);
    // vm stored transposed so it is the column-major B of stage 3.
    dim3 g1(D / 128, N / 128);
    gemm_bf16_wmma<OUT_BF16_ROW  ><<<g1, 256, 0, stream>>>(xb, qT, qmb, N, D, D, D, 1.0f);
    gemm_bf16_wmma<OUT_BF16_ROW  ><<<g1, 256, 0, stream>>>(xb, kT, kmb, N, D, D, D, 1.0f);
    gemm_bf16_wmma<OUT_BF16_TRANS><<<g1, 256, 0, stream>>>(xb, vT, vmT, N, D, D, N, 1.0f);

    // Stage 2: scores = (qm @ km.T) / sqrt(1024)   [N x N], bf16
    dim3 g2(N / 128, N / 128);
    gemm_bf16_wmma<OUT_BF16_ROW><<<g2, 256, 0, stream>>>(qmb, kmb, sb, N, N, D, N, 0.03125f);

    // Stage 3: s = scores @ vm   [N x D], f32 straight into d_out
    dim3 g3(D / 128, N / 128);
    gemm_bf16_wmma<OUT_F32_ROW><<<g3, 256, 0, stream>>>(sb, vmT, out, N, D, N, D, 1.0f);

    // Stage 4: softmax over last dim, in place on d_out
    softmax_rows_1024<<<N, 256, 0, stream>>>(out);
}
